// Attention_8761733284371
// MI455X (gfx1250) — matrix operand order repaired and checked by reading, not compiled, not run on hardware
//
#include <hip/hip_runtime.h>
#include <hip/hip_bf16.h>

// CDNA5 / gfx1250, wave32. Matrix math via v_wmma_f32_16x16x32_f16.
typedef __attribute__((ext_vector_type(16))) _Float16 v16h;
typedef __attribute__((ext_vector_type(8)))  _Float16 v8h;
typedef __attribute__((ext_vector_type(8)))  float    v8f;

#define CDIM 128
#define NPIX 4096   // 64*64
#define MPIX 1024   // 32*32 (pooled)

// ---------------------------------------------------------------------------
// Fragment index helpers (ISA 7.12.2, wave32):
//  A (16xK f16): lane%16 = M row; K = e + 16*h (this code's order, not the ISA table's).
//  B (Kx16 f16): lane%16 = N col; K = e + 16*h (this code's order, not the ISA table's).
//  C/D (16x16 f32): VGPR r -> M = r + 8*h, N = lane%16.   (h = lane/16)
// ---------------------------------------------------------------------------
__device__ __forceinline__ int a_koff(int e, int h) {
  return e + 16 * h;
}
__device__ __forceinline__ int b_koff(int e, int h) {
  return e + 16 * h;
}
__device__ __forceinline__ v8f wmma_f16(const v16h& a, const v16h& b, const v8f& c) {
  return __builtin_amdgcn_wmma_f32_16x16x32_f16(false, a, false, b, (short)0, c,
                                                false, false);
}

// ---------------------------------------------------------------------------
// Kernel 0: one-shot f32 -> f16 conversion of the four weight matrices.
// dst: [0,2048) Wt | [2048,4096) Wp | [4096,12288) Wg | [12288,20480) Wo
// ---------------------------------------------------------------------------
__global__ void wcvt_kernel(const float* __restrict__ Wt,
                            const float* __restrict__ Wp,
                            const float* __restrict__ Wg,
                            const float* __restrict__ Wo,
                            _Float16* __restrict__ dst) {
  int i = blockIdx.x * blockDim.x + threadIdx.x;
  if (i >= 20480) return;
  float v;
  if (i < 2048)        v = Wt[i];
  else if (i < 4096)   v = Wp[i - 2048];
  else if (i < 12288)  v = Wg[i - 4096];
  else                 v = Wo[i - 12288];
  dst[i] = (_Float16)v;
}

// ---------------------------------------------------------------------------
// Kernel 1: 1x1-conv projections theta/phi/g via WMMA.
// Block = 256 thr = 8 waves; each wave owns one 16-position tile staged into
// LDS with async global->LDS copies. Grid = B * 32.
// theta stored TRANSPOSED [b][pos][16]; phi/g channel-major for pooling.
// ---------------------------------------------------------------------------
__global__ void proj_kernel(const float* __restrict__ x,
                            const _Float16* __restrict__ W16,
                            _Float16* __restrict__ thetaT,
                            _Float16* __restrict__ phiF,
                            _Float16* __restrict__ gF) {
  extern __shared__ char smem[];
  float* xs = (float*)smem;                 // [8 waves][128 ch][16 pos] f32
  const int tid  = threadIdx.x;
  const int wave = tid >> 5;                // wave32
  const int lane = tid & 31;
  const int b    = blockIdx.x >> 5;         // 32 blocks per batch
  const int blk  = blockIdx.x & 31;
  const int pos0 = blk * 128 + wave * 16;
  const float* xb = x + (size_t)b * CDIM * NPIX;

  float* xw = xs + wave * (CDIM * 16);
  const unsigned long long xbase = (unsigned long long)xb;
  const unsigned xw_lds = (unsigned)(unsigned long long)(size_t)xw;

  // async global->LDS staging: 512 x 16B per wave (8 KB tile), 16 per lane
#pragma unroll
  for (int i = 0; i < 16; ++i) {
    int idx  = lane + 32 * i;               // 0..511
    int row  = idx >> 2;                    // channel
    int quad = idx & 3;
    unsigned voff = (unsigned)(((size_t)row * NPIX + pos0 + quad * 4) * 4);
    unsigned ldsa = xw_lds + (unsigned)((row * 16 + quad * 4) * 4);
    asm volatile("global_load_async_to_lds_b128 %0, %1, %2 offset:0"
                 :: "v"(ldsa), "v"(voff), "s"(xbase)
                 : "memory");
  }
  asm volatile("s_wait_asynccnt 0" ::: "memory");
  __syncthreads();

  const int h    = lane >> 4;
  const int mrow = lane & 15;               // A-row / B-col / D-col index

  // B fragments (shared by all three projections): x-tile, 4 K-chunks of 32
  v16h bfrag[4];
#pragma unroll
  for (int kc = 0; kc < 4; ++kc) {
#pragma unroll
    for (int e = 0; e < 16; ++e) {
      int k = kc * 32 + b_koff(e, h);
      bfrag[kc][e] = (_Float16)xw[k * 16 + mrow];
    }
  }

  const _Float16* Wt16 = W16;
  const _Float16* Wp16 = W16 + 2048;
  const _Float16* Wg16 = W16 + 4096;

  // ---- theta (16 out-ch), store transposed [pos][16] ----
  {
    v8f acc;
#pragma unroll
    for (int r = 0; r < 8; ++r) acc[r] = 0.0f;
#pragma unroll
    for (int kc = 0; kc < 4; ++kc) {
      v16h a;
#pragma unroll
      for (int e = 0; e < 16; ++e)
        a[e] = Wt16[mrow * CDIM + kc * 32 + a_koff(e, h)];
      acc = wmma_f16(a, bfrag[kc], acc);
    }
    v8h th;
#pragma unroll
    for (int r = 0; r < 8; ++r) th[r] = (_Float16)acc[r];
    _Float16* dst = thetaT + (size_t)b * NPIX * 16;
    *(v8h*)(dst + (size_t)(pos0 + mrow) * 16 + 8 * h) = th;   // 16B packed store
  }

  // ---- phi (16 out-ch) + g (64 out-ch), channel-major for pooling ----
#pragma unroll
  for (int pg = 0; pg < 5; ++pg) {         // pg==0: phi ; pg 1..4: g m-tiles
    const _Float16* W = (pg == 0) ? Wp16 : (Wg16 + (pg - 1) * 16 * CDIM);
    _Float16* dst = (pg == 0) ? (phiF + (size_t)b * 16 * NPIX)
                              : (gF + (size_t)b * 64 * NPIX + (size_t)(pg - 1) * 16 * NPIX);
    v8f acc;
#pragma unroll
    for (int r = 0; r < 8; ++r) acc[r] = 0.0f;
#pragma unroll
    for (int kc = 0; kc < 4; ++kc) {
      v16h a;
#pragma unroll
      for (int e = 0; e < 16; ++e)
        a[e] = W[mrow * CDIM + kc * 32 + a_koff(e, h)];
      acc = wmma_f16(a, bfrag[kc], acc);
    }
#pragma unroll
    for (int r = 0; r < 8; ++r)
      dst[(size_t)(r + 8 * h) * NPIX + pos0 + mrow] = (_Float16)acc[r];
  }
}

// ---------------------------------------------------------------------------
// Kernel 2: 2x2 spatial maxpool. phi -> TRANSPOSED [b][m][16]; g -> [b][c][m].
// ---------------------------------------------------------------------------
__global__ void pool_kernel(const _Float16* __restrict__ phiF,
                            const _Float16* __restrict__ gF,
                            _Float16* __restrict__ phiPT,
                            _Float16* __restrict__ gP) {
  const int TOT = 8 * (16 + 64) * MPIX;
  int i = blockIdx.x * blockDim.x + threadIdx.x;
  if (i >= TOT) return;
  int mp = i & (MPIX - 1);
  int bc = i >> 10;                          // b*80 + ch
  int b  = bc / 80;
  int ch = bc % 80;
  const _Float16* src;
  if (ch < 16) src = phiF + ((size_t)b * 16 + ch) * NPIX;
  else         src = gF + ((size_t)b * 64 + (ch - 16)) * NPIX;
  int hp = mp >> 5, wp = mp & 31;
  int p00 = (2 * hp) * 64 + 2 * wp;
  float a0 = (float)src[p00],      a1 = (float)src[p00 + 1];
  float a2 = (float)src[p00 + 64], a3 = (float)src[p00 + 65];
  _Float16 m = (_Float16)fmaxf(fmaxf(a0, a1), fmaxf(a2, a3));
  if (ch < 16) phiPT[((size_t)b * MPIX + mp) * 16 + ch] = m;   // transposed
  else         gP[((size_t)b * 64 + (ch - 16)) * MPIX + mp] = m;
}

// ---------------------------------------------------------------------------
// Kernel 3: fused attention. One block (256 thr = 8 waves) per 16-query tile.
// ---------------------------------------------------------------------------
__global__ void attn_kernel(const float* __restrict__ x,
                            const _Float16* __restrict__ thetaT,
                            const _Float16* __restrict__ phiPT,
                            const _Float16* __restrict__ gP,
                            const _Float16* __restrict__ Wo16,
                            const float* __restrict__ gammaPtr,
                            float* __restrict__ out) {
  extern __shared__ char smem[];
  float*    sS   = (float*)smem;                            // 16*1024 f32
  _Float16* sP   = (_Float16*)(smem + 65536);               // 16*1024 f16
  float*    sO   = (float*)(smem + 65536 + 32768);          // 2*16*64 f32
  float*    sRed = (float*)(smem + 65536 + 32768 + 8192);   // 256 f32
  float*    sStat = sRed + 256;                             // 16 max + 16 sum

  const int tid  = threadIdx.x;
  const int wave = tid >> 5;
  const int lane = tid & 31;
  const int h    = lane >> 4;
  const int mrow = lane & 15;
  const int b    = blockIdx.x >> 8;     // 256 query tiles per batch
  const int qt   = blockIdx.x & 255;
  const int n0   = qt * 16;

  const _Float16* thb = thetaT + (size_t)b * NPIX * 16;
  const _Float16* phb = phiPT  + (size_t)b * MPIX * 16;
  const _Float16* gb  = gP     + (size_t)b * 64 * MPIX;

  // prefetch this wave's g panel into cache (global_prefetch_b8)
  __builtin_prefetch(gb + (size_t)wave * 8 * MPIX, 0, 3);

  // --- A fragment: 16 queries x K32 (K 16..31 zero-padded); lanes 0-15 read one 32B run ---
  v16h aTh;
#pragma unroll
  for (int e = 0; e < 16; ++e) {
    int k = a_koff(e, h);
    aTh[e] = (k < 16) ? thb[(size_t)(n0 + mrow) * 16 + k] : (_Float16)0.0f;
  }

  // --- Phase A: scores; each wave fills 8 column tiles of 16 ---
#pragma unroll
  for (int c8 = 0; c8 < 8; ++c8) {
    int m0 = (wave * 8 + c8) * 16;
    v16h bPhi;
#pragma unroll
    for (int e = 0; e < 16; ++e) {
      int k = b_koff(e, h);
      bPhi[e] = (k < 16) ? phb[(size_t)(m0 + mrow) * 16 + k] : (_Float16)0.0f;
    }
    v8f acc;
#pragma unroll
    for (int r = 0; r < 8; ++r) acc[r] = 0.0f;
    acc = wmma_f16(aTh, bPhi, acc);
#pragma unroll
    for (int r = 0; r < 8; ++r)
      sS[(r + 8 * h) * MPIX + m0 + mrow] = acc[r];
  }
  __syncthreads();

  // --- Phase B: softmax over 1024 per row; 16 threads per row ---
  const int row = tid >> 4;
  const int sub = tid & 15;
  float lm = -3.4e38f;
  for (int i = 0; i < 64; ++i) lm = fmaxf(lm, sS[row * MPIX + sub + 16 * i]);
  sRed[tid] = lm;
  __syncthreads();
  if (sub == 0) {
    float m = sRed[row * 16];
    for (int t = 1; t < 16; ++t) m = fmaxf(m, sRed[row * 16 + t]);
    sStat[row] = m;
  }
  __syncthreads();
  const float rmax = sStat[row];
  float ls = 0.0f;
  for (int i = 0; i < 64; ++i) {
    float e = __expf(sS[row * MPIX + sub + 16 * i] - rmax);
    sS[row * MPIX + sub + 16 * i] = e;
    ls += e;
  }
  sRed[tid] = ls;
  __syncthreads();
  if (sub == 0) {
    float s = 0.0f;
    for (int t = 0; t < 16; ++t) s += sRed[row * 16 + t];
    sStat[16 + row] = s;
  }
  __syncthreads();
  const float rinv = 1.0f / sStat[16 + row];
  for (int i = 0; i < 64; ++i)
    sP[row * MPIX + sub + 16 * i] =
        (_Float16)(sS[row * MPIX + sub + 16 * i] * rinv);
  __syncthreads();

  // --- Phase C: O[n][c] = sum_m P[n][m] * g[c][m]; K=1024 split in halves ---
  {
    const int ctile = wave & 3;        // c-tile 0..3 (64 channels)
    const int khalf = wave >> 2;       // K half 0/1
    const int c0 = ctile * 16;
    v8f acc;
#pragma unroll
    for (int r = 0; r < 8; ++r) acc[r] = 0.0f;
#pragma unroll 4
    for (int kk = 0; kk < 16; ++kk) {
      int k0 = (khalf * 16 + kk) * 32;
      v16h aP, bG;
#pragma unroll
      for (int e = 0; e < 16; ++e) {
        aP[e] = sP[mrow * MPIX + k0 + a_koff(e, h)];
        bG[e] = gb[(size_t)(c0 + mrow) * MPIX + k0 + b_koff(e, h)];
      }
      acc = wmma_f16(aP, bG, acc);
    }
#pragma unroll
    for (int r = 0; r < 8; ++r)
      sO[(khalf * 16 + r + 8 * h) * 64 + c0 + mrow] = acc[r];
  }
  __syncthreads();
#pragma unroll
  for (int i = 0; i < 4; ++i) {
    int idx = tid + 256 * i;
    sO[idx] += sO[1024 + idx];
  }
  __syncthreads();

  // --- Phase D: out = gamma * (W_o x O) + x ; wave -> 16 output channels ---
  {
    const int oc0 = wave * 16;
    v8f acc;
#pragma unroll
    for (int r = 0; r < 8; ++r) acc[r] = 0.0f;
#pragma unroll
    for (int kc = 0; kc < 2; ++kc) {   // K = 64
      v16h aW, bO;
#pragma unroll
      for (int e = 0; e < 16; ++e) {
        aW[e] = Wo16[(size_t)(oc0 + mrow) * 64 + kc * 32 + a_koff(e, h)];
        bO[e] = (_Float16)sO[mrow * 64 + kc * 32 + b_koff(e, h)];
      }
      acc = wmma_f16(aW, bO, acc);
    }
    const float gamma = gammaPtr[0];
    const float* xb = x + (size_t)b * CDIM * NPIX;
    float* ob = out + (size_t)b * CDIM * NPIX;
#pragma unroll
    for (int r = 0; r < 8; ++r) {
      int oc = oc0 + r + 8 * h;
      size_t off = (size_t)oc * NPIX + n0 + mrow;
      ob[off] = gamma * acc[r] + xb[off];
    }
  }
}

// ---------------------------------------------------------------------------
extern "C" void kernel_launch(void* const* d_in, const int* in_sizes, int n_in,
                              void* d_out, int out_size, void* d_ws,
                              size_t ws_size, hipStream_t stream) {
  const float* x     = (const float*)d_in[0];
  const float* Wt    = (const float*)d_in[1];
  const float* Wp    = (const float*)d_in[2];
  const float* Wg    = (const float*)d_in[3];
  const float* Wo    = (const float*)d_in[4];
  const float* gamma = (const float*)d_in[5];
  float* out = (float*)d_out;

  // workspace carve-up (f16 intermediates, ~7.3 MB total)
  char* ws = (char*)d_ws;
  _Float16* thetaT = (_Float16*)ws;  ws += (size_t)8 * NPIX * 16 * 2;  // [b][pos][16]
  _Float16* phiF   = (_Float16*)ws;  ws += (size_t)8 * 16 * NPIX * 2;  // [b][ch][pos]
  _Float16* gF     = (_Float16*)ws;  ws += (size_t)8 * 64 * NPIX * 2;  // [b][ch][pos]
  _Float16* phiPT  = (_Float16*)ws;  ws += (size_t)8 * MPIX * 16 * 2;  // [b][m][16]
  _Float16* gP     = (_Float16*)ws;  ws += (size_t)8 * 64 * MPIX * 2;  // [b][c][m]
  _Float16* W16    = (_Float16*)ws;  ws += (size_t)20480 * 2;          // weights f16

  wcvt_kernel<<<(20480 + 255) / 256, 256, 0, stream>>>(Wt, Wp, Wg, Wo, W16);

  proj_kernel<<<8 * 32, 256, 64 * 1024, stream>>>(x, W16, thetaT, phiF, gF);

  const int poolTot = 8 * (16 + 64) * MPIX;
  pool_kernel<<<(poolTot + 255) / 256, 256, 0, stream>>>(phiF, gF, phiPT, gP);

  const size_t lds3 = 65536 + 32768 + 8192 + 1024 + 128;
  attn_kernel<<<8 * 256, 256, lds3, stream>>>(x, thetaT, phiPT, gP,
                                              W16 + 12288, gamma, out);
}
